// TransformerAttLayer_2576980378153
// MI455X (gfx1250) — matrix operand order repaired and checked by reading, not compiled, not run on hardware
//
#include <hip/hip_runtime.h>
#include <hip/hip_bf16.h>

typedef __bf16 bf16_t;
typedef __bf16 v16bf __attribute__((ext_vector_type(16)));
typedef float  v8f   __attribute__((ext_vector_type(8)));

#define N_NODES 30000
#define N_EDGES 15000
#define DEG_V 16
#define DEG_E 10
#define SSTR 136   /* LDS row stride (bf16 elems) for 128-wide buffers (16B-aligned rows) */
#define KSTR 264   /* LDS row stride for 256-wide ke overlay */

struct alignas(16) U128 { unsigned int w[4]; };
union Frag { v16bf v; U128 u[2]; };

// ---------------- parameter structs ----------------
struct IsabP {
  const float*  qp0;                       // [4][128] precomputed I@wq+bq
  const bf16_t *wk0t, *wv0t, *wo0t;        // transposed bf16 [N=128][K]
  const float  *bk0, *bv0, *bo0;
  const bf16_t *wq1t, *wk1t, *wv1t, *wo1t;
  const float  *bq1, *bk1, *bv1, *bo1;
};
struct StageParams {
  const float* srcfeat;   // gather source [*,64]
  const int*   nbr;       // [B][S]
  int          S;
  IsabP        isab0, isab1;
  const bf16_t* fc_kt;    // [256][128] transposed bf16
  const float*  g;        // [4][64]  folded fc_q @ attn_q
  const float*  qfeat;    // [B][64]
  const float*  attn_k;   // [4][64]
  const float*  bias;     // [256]
  float*        out;      // [B][64]
};

// ---------------- prep kernels ----------------
__global__ void krn_transpose_bf16(const float* __restrict__ src, bf16_t* __restrict__ dst,
                                   int r, int c) {
  int t = blockIdx.x * blockDim.x + threadIdx.x;
  if (t >= r * c) return;
  int i = t / c, j = t % c;
  dst[(size_t)j * r + i] = (bf16_t)src[(size_t)i * c + j];
}

__global__ void krn_qp0(const float* __restrict__ I, const float* __restrict__ wq,
                        const float* __restrict__ bq, float* __restrict__ qp0) {
  int t = blockIdx.x * blockDim.x + threadIdx.x;
  if (t >= 512) return;
  int i = t >> 7, c = t & 127;
  float acc = bq[c];
  for (int k = 0; k < 128; ++k) acc += I[i * 128 + k] * wq[k * 128 + c];
  qp0[t] = acc;
}

__global__ void krn_g(const float* __restrict__ fcq, const float* __restrict__ attnq,
                      float* __restrict__ g) {
  int t = blockIdx.x * blockDim.x + threadIdx.x;
  if (t >= 256) return;
  int h = t >> 6, k = t & 63;
  float acc = 0.0f;
  for (int c = 0; c < 64; ++c) acc += fcq[k * 256 + h * 64 + c] * attnq[h * 64 + c];
  g[h * 64 + k] = acc;
}

// ---------------- wave-level GEMM: C[16,N] = A[16,K] @ W + bias (epilogue) ----------------
// k-outer / n-inner: each A fragment is loaded from LDS once and reused across all
// N/16 column tiles (register-blocked accumulators, 8x fewer DS loads than n-outer).
enum { EPI_STORE = 0, EPI_RELU_ADD = 1 };

template <int K, int N, int EPI>
__device__ __forceinline__ void wave_gemm(const bf16_t* Asm, const bf16_t* __restrict__ Wt,
                                          const float* __restrict__ bias,
                                          bf16_t* Dsm, int dstride, const bf16_t* Srcsm,
                                          int lane) {
  static_assert(N % 16 == 0 && N <= 128, "N tiles");
  constexpr int NT = N / 16;
  const int hi = lane >> 4, lo = lane & 15;

  v8f acc[NT];
#pragma unroll
  for (int n = 0; n < NT; ++n) acc[n] = {};

#pragma unroll
  for (int k0 = 0; k0 < K; k0 += 32) {
    // A (16x32 bf16): lane = row m; elems 0..7 -> K = k0+hi*8 .. +7 ; 8..15 -> +16..+23
    Frag a;
    const bf16_t* ap = Asm + lo * SSTR + k0 + hi * 8;
    a.u[0] = *(const U128*)(ap);
    a.u[1] = *(const U128*)(ap + 16);
#pragma unroll
    for (int n = 0; n < NT; ++n) {
      // B (32x16 bf16): lane = col; elems 0..7 -> K = k0+hi*8 .. +7 ; 8..15 -> +16..+23 (Wt is [N][K])
      Frag b;
      const bf16_t* bp = Wt + (size_t)(n * 16 + lo) * K + k0 + hi * 8;
      b.u[0] = *(const U128*)(bp);
      b.u[1] = *(const U128*)(bp + 16);
      acc[n] = __builtin_amdgcn_wmma_f32_16x16x32_bf16(false, a.v, false, b.v,
                                                       (short)0, acc[n], false, false);
    }
  }

#pragma unroll
  for (int n = 0; n < NT; ++n) {
    float bv = bias ? bias[n * 16 + lo] : 0.0f;
#pragma unroll
    for (int r = 0; r < 8; ++r) {
      int m = r + hi * 8;                 // C layout: VGPR r -> row r (+8 for lanes 16..31)
      float v = acc[n][r] + bv;
      if (EPI == EPI_RELU_ADD) v = (float)Srcsm[m * SSTR + n * 16 + lo] + fmaxf(v, 0.0f);
      Dsm[m * dstride + n * 16 + lo] = (bf16_t)v;
    }
  }
}

// ---------------- attention (VALU, tiny) ----------------
__device__ __forceinline__ void attn_mab0(const float* __restrict__ qp0, const bf16_t* Ksm,
                                          const bf16_t* Vsm, int S, bf16_t* Osm,
                                          float* fscr, int lane) {
  const float scale = 0.088388347648318447f; // 1/sqrt(128)
  for (int t = lane; t < 256; t += 32) {     // t = h*64 + i*16 + j
    int j = t & 15, i = (t >> 4) & 3, h = t >> 6;
    float s;
    if (j < S) {
      s = 0.0f;
#pragma unroll
      for (int c = 0; c < 32; ++c)
        s += qp0[i * 128 + h * 32 + c] * (float)Ksm[j * SSTR + h * 32 + c];
      s *= scale;
    } else s = -1e30f;
    fscr[t] = s;
  }
  __syncthreads();
  if (lane < 16) {                           // softmax over j per (h,i)
    int base = lane * 16;
    float m = -1e30f;
    for (int j = 0; j < 16; ++j) m = fmaxf(m, fscr[base + j]);
    float sum = 0.0f;
    for (int j = 0; j < 16; ++j) { float e = __expf(fscr[base + j] - m); fscr[base + j] = e; sum += e; }
    float inv = 1.0f / sum;
    for (int j = 0; j < 16; ++j) fscr[base + j] *= inv;
  }
  __syncthreads();
  for (int t = lane; t < 512; t += 32) {     // O rows 0..3
    int i = t >> 7, c = t & 127, h = c >> 5;
    float o = 0.0f;
    for (int j = 0; j < 16; ++j)
      o += fscr[h * 64 + i * 16 + j] * (float)Vsm[j * SSTR + c];
    Osm[i * SSTR + c] = (bf16_t)(o + qp0[i * 128 + c]);
  }
  for (int t = lane; t < 12 * 128; t += 32) { // zero rows 4..15
    int i = 4 + (t >> 7), c = t & 127;
    Osm[i * SSTR + c] = (bf16_t)0.0f;
  }
}

__device__ __forceinline__ void attn_mab1(const bf16_t* Qsm, const bf16_t* Ksm,
                                          const bf16_t* Vsm, bf16_t* Osm,
                                          float* fscr, int lane) {
  const float scale = 0.088388347648318447f;
  for (int t = lane; t < 256; t += 32) {     // t = h*64 + i*4 + j
    int j = t & 3, i = (t >> 2) & 15, h = t >> 6;
    float s = 0.0f;
#pragma unroll
    for (int c = 0; c < 32; ++c)
      s += (float)Qsm[i * SSTR + h * 32 + c] * (float)Ksm[j * SSTR + h * 32 + c];
    fscr[t] = s * scale;
  }
  __syncthreads();
  for (int r = lane; r < 64; r += 32) {      // softmax over 4 keys
    int base = r * 4;
    float a0 = fscr[base], a1 = fscr[base + 1], a2 = fscr[base + 2], a3 = fscr[base + 3];
    float m = fmaxf(fmaxf(a0, a1), fmaxf(a2, a3));
    float e0 = __expf(a0 - m), e1 = __expf(a1 - m), e2 = __expf(a2 - m), e3 = __expf(a3 - m);
    float inv = 1.0f / (e0 + e1 + e2 + e3);
    fscr[base] = e0 * inv; fscr[base + 1] = e1 * inv;
    fscr[base + 2] = e2 * inv; fscr[base + 3] = e3 * inv;
  }
  __syncthreads();
  for (int t = lane; t < 2048; t += 32) {    // O2 = Qp2 + A@V, all 16 rows
    int i = t >> 7, c = t & 127, h = c >> 5;
    float o = 0.0f;
#pragma unroll
    for (int j = 0; j < 4; ++j)
      o += fscr[h * 64 + i * 4 + j] * (float)Vsm[j * SSTR + c];
    Osm[i * SSTR + c] = (bf16_t)(o + (float)Qsm[i * SSTR + c]);
  }
}

// ---------------- one ISAB block (in-wave) ----------------
template <int K>
__device__ void isab_layer(const IsabP& ip, bf16_t*& X, bf16_t*& T1, bf16_t*& T2, bf16_t*& T3,
                           float* fscr, int S, int lane) {
  wave_gemm<K, 128, EPI_STORE>(X, ip.wk0t, ip.bk0, T1, SSTR, nullptr, lane);   // Kp
  wave_gemm<K, 128, EPI_STORE>(X, ip.wv0t, ip.bv0, T2, SSTR, nullptr, lane);   // Vp
  __syncthreads();
  attn_mab0(ip.qp0, T1, T2, S, T3, fscr, lane);                                // O (rows 0..3)
  __syncthreads();
  wave_gemm<128, 128, EPI_RELU_ADD>(T3, ip.wo0t, ip.bo0, T1, SSTR, T3, lane);  // H = O + relu(O@wo+bo)
  wave_gemm<K, 128, EPI_STORE>(X, ip.wq1t, ip.bq1, T2, SSTR, nullptr, lane);   // Qp2
  __syncthreads();
  wave_gemm<128, 128, EPI_STORE>(T1, ip.wk1t, ip.bk1, T3, SSTR, nullptr, lane); // Kp2
  wave_gemm<128, 128, EPI_STORE>(T1, ip.wv1t, ip.bv1, X, SSTR, nullptr, lane);  // Vp2 (X dead)
  __syncthreads();
  attn_mab1(T2, T3, X, T1, fscr, lane);                                        // O2 = Qp2 + A@Vp2
  __syncthreads();
  wave_gemm<128, 128, EPI_RELU_ADD>(T1, ip.wo1t, ip.bo1, T2, SSTR, T1, lane);  // newX
  __syncthreads();
  bf16_t* nX = T2; T2 = X; X = nX;
}

// ---------------- stage kernel: one set per wave ----------------
__global__ __launch_bounds__(32) void stage_kernel(StageParams P) {
  __shared__ __align__(16) bf16_t sm[4 * 16 * SSTR];
  __shared__ float fscr[320];
  const int lane = threadIdx.x;
  const int b = blockIdx.x;
  const int S = P.S;

  bf16_t* X  = sm;
  bf16_t* T1 = sm + 16 * SSTR;
  bf16_t* T2 = sm + 32 * SSTR;
  bf16_t* T3 = sm + 48 * SSTR;

  for (int i = lane; i < 4 * 16 * SSTR; i += 32) sm[i] = (bf16_t)0.0f;
  __syncthreads();

  // gather set rows
  for (int t = lane; t < S * 64; t += 32) {
    int s = t >> 6, c = t & 63;
    int idx = P.nbr[b * S + s];
    X[s * SSTR + c] = (bf16_t)P.srcfeat[(size_t)idx * 64 + c];
  }
  __syncthreads();

  isab_layer<64>(P.isab0, X, T1, T2, T3, fscr, S, lane);
  isab_layer<128>(P.isab1, X, T1, T2, T3, fscr, S, lane);

  // ke = X @ fc_k -> [16][256], overlaid on the dead ping-pong buffers
  // (after two swaps X sits at one quarter; the other three quarters are free
  //  and 16*KSTR bf16 = 8448B fits in two adjacent free quarters).
  bf16_t* kesm = (X == sm) ? (sm + 16 * SSTR) : sm;
  wave_gemm<128, 128, EPI_STORE>(X, P.fc_kt,             nullptr, kesm,       KSTR, nullptr, lane);
  wave_gemm<128, 128, EPI_STORE>(X, P.fc_kt + 128 * 128, nullptr, kesm + 128, KSTR, nullptr, lane);
  __syncthreads();

  // el[s][h]
  for (int t = lane; t < 64; t += 32) {
    int s = t >> 2, h = t & 3;
    float e = 0.0f;
    for (int c = 0; c < 64; ++c)
      e += (float)kesm[s * KSTR + h * 64 + c] * P.attn_k[h * 64 + c];
    fscr[t] = e;
  }
  // er[h] = qfeat[b] . g[h]
  if (lane < 4) {
    float e = 0.0f;
    for (int k = 0; k < 64; ++k) e += P.qfeat[(size_t)b * 64 + k] * P.g[lane * 64 + k];
    fscr[64 + lane] = e;
  }
  __syncthreads();
  // a = softmax_s relu(el + er)
  if (lane < 4) {
    int h = lane;
    float er = fscr[64 + h];
    float vv[16];
    float m = -1e30f;
    for (int s = 0; s < 16; ++s) {
      float x = (s < S) ? fmaxf(fscr[s * 4 + h] + er, 0.0f) : -1e30f;
      vv[s] = x; m = fmaxf(m, x);
    }
    float sum = 0.0f;
    for (int s = 0; s < 16; ++s) {
      float e = (s < S) ? __expf(vv[s] - m) : 0.0f;
      vv[s] = e; sum += e;
    }
    float inv = 1.0f / sum;
    for (int s = 0; s < 16; ++s) fscr[s * 4 + h] = vv[s] * inv;
  }
  __syncthreads();
  // out[c] = mean_h ( sum_s a*ke + bias )
  for (int t = lane; t < 64; t += 32) {
    int c = t;
    float acc = 0.0f;
    for (int h = 0; h < 4; ++h) {
      float s_acc = 0.0f;
      for (int s = 0; s < S; ++s)
        s_acc += fscr[s * 4 + h] * (float)kesm[s * KSTR + h * 64 + c];
      acc += s_acc + P.bias[h * 64 + c];
    }
    P.out[(size_t)b * 64 + c] = 0.25f * acc;
  }
}

// ---------------- host launch ----------------
extern "C" void kernel_launch(void* const* d_in, const int* in_sizes, int n_in,
                              void* d_out, int out_size, void* d_ws, size_t ws_size,
                              hipStream_t stream) {
  (void)in_sizes; (void)out_size; (void)ws_size;
  if (n_in < 82) return;

  const float* vfeat = (const float*)d_in[0];
  const float* efeat = (const float*)d_in[1];
  const int*   nbr_v = (const int*)d_in[2];
  const int*   nbr_e = (const int*)d_in[3];

  // per-ISAB leaf base indices (dict insertion order) and input dims
  const int LB[4]  = {4, 21, 38, 55};   // enc_v[0], enc_v[1], enc_e[0], enc_e[1]
  const int DIN[4] = {64, 128, 64, 128};

  char* ws = (char*)d_ws;
  size_t cur = 0;
  auto alloc = [&](size_t bytes) -> char* {
    char* p = ws + cur;
    cur = (cur + bytes + 255) & ~(size_t)255;
    return p;
  };

  bf16_t* wT[4][7];
  float*  qp0[4];
  for (int l = 0; l < 4; ++l) {
    int din = DIN[l];
    const int srcIdx[7] = {LB[l] + 3, LB[l] + 5, LB[l] + 7, LB[l] + 9,
                           LB[l] + 11, LB[l] + 13, LB[l] + 15};  // wk0,wv0,wo0,wq1,wk1,wv1,wo1
    const int rows[7] = {din, din, 128, din, 128, 128, 128};
    for (int m = 0; m < 7; ++m) {
      int r = rows[m], c = 128;
      wT[l][m] = (bf16_t*)alloc((size_t)r * c * sizeof(bf16_t));
      int total = r * c;
      krn_transpose_bf16<<<(total + 255) / 256, 256, 0, stream>>>(
          (const float*)d_in[srcIdx[m]], wT[l][m], r, c);
    }
    qp0[l] = (float*)alloc(512 * sizeof(float));
    krn_qp0<<<2, 256, 0, stream>>>((const float*)d_in[LB[l] + 0],
                                   (const float*)d_in[LB[l] + 1],
                                   (const float*)d_in[LB[l] + 2], qp0[l]);
  }
  bf16_t* fc_ke_t = (bf16_t*)alloc((size_t)128 * 256 * sizeof(bf16_t));
  krn_transpose_bf16<<<(128 * 256 + 255) / 256, 256, 0, stream>>>(
      (const float*)d_in[72], fc_ke_t, 128, 256);
  bf16_t* fc_kv_t = (bf16_t*)alloc((size_t)128 * 256 * sizeof(bf16_t));
  krn_transpose_bf16<<<(128 * 256 + 255) / 256, 256, 0, stream>>>(
      (const float*)d_in[77], fc_kv_t, 128, 256);
  float* g_e = (float*)alloc(256 * sizeof(float));
  krn_g<<<1, 256, 0, stream>>>((const float*)d_in[73], (const float*)d_in[75], g_e);
  float* g_v = (float*)alloc(256 * sizeof(float));
  krn_g<<<1, 256, 0, stream>>>((const float*)d_in[78], (const float*)d_in[80], g_v);

  float* out_v = (float*)d_out;
  float* out_e = out_v + (size_t)N_NODES * 64;

  auto mkIsab = [&](int l) -> IsabP {
    IsabP p;
    p.qp0 = qp0[l];
    p.wk0t = wT[l][0]; p.wv0t = wT[l][1]; p.wo0t = wT[l][2];
    p.wq1t = wT[l][3]; p.wk1t = wT[l][4]; p.wv1t = wT[l][5]; p.wo1t = wT[l][6];
    p.bk0 = (const float*)d_in[LB[l] + 4];
    p.bv0 = (const float*)d_in[LB[l] + 6];
    p.bo0 = (const float*)d_in[LB[l] + 8];
    p.bq1 = (const float*)d_in[LB[l] + 10];
    p.bk1 = (const float*)d_in[LB[l] + 12];
    p.bv1 = (const float*)d_in[LB[l] + 14];
    p.bo1 = (const float*)d_in[LB[l] + 16];
    return p;
  };

  // stage 1: node -> hyperedge
  StageParams P1;
  P1.srcfeat = vfeat; P1.nbr = nbr_v; P1.S = DEG_V;
  P1.isab0 = mkIsab(0); P1.isab1 = mkIsab(1);
  P1.fc_kt = fc_ke_t; P1.g = g_e; P1.qfeat = efeat;
  P1.attn_k = (const float*)d_in[74]; P1.bias = (const float*)d_in[76];
  P1.out = out_e;
  stage_kernel<<<N_EDGES, 32, 0, stream>>>(P1);

  // stage 2: hyperedge -> node (consumes out_e)
  StageParams P2;
  P2.srcfeat = out_e; P2.nbr = nbr_e; P2.S = DEG_E;
  P2.isab0 = mkIsab(2); P2.isab1 = mkIsab(3);
  P2.fc_kt = fc_kv_t; P2.g = g_v; P2.qfeat = vfeat;
  P2.attn_k = (const float*)d_in[79]; P2.bias = (const float*)d_in[81];
  P2.out = out_v;
  stage_kernel<<<N_NODES, 32, 0, stream>>>(P2);
}
